// GroupedQueryAttention_76605036691519
// MI455X (gfx1250) — matrix operand order repaired and checked by reading, not compiled, not run on hardware
//
#include <hip/hip_runtime.h>
#include <math.h>

typedef __bf16 bf16;
typedef __attribute__((ext_vector_type(16))) __bf16 v16bf;
typedef __attribute__((ext_vector_type(8)))  __bf16 v8bf;
typedef __attribute__((ext_vector_type(8)))  float  v8f;
typedef __attribute__((ext_vector_type(4)))  float  v4f;
typedef __attribute__((ext_vector_type(4)))  int    v4i;

#define LSEQ   2048
#define DMODEL 2048
#define NH     32
#define NKV    8
#define DKH    64
// G = NH/NKV = 4

#define ASG __attribute__((address_space(1)))
#define ASL __attribute__((address_space(3)))

#if __has_builtin(__builtin_amdgcn_global_load_async_to_lds_b128)
#define ASYNC_COPY 1
#else
#define ASYNC_COPY 0
#endif

__device__ __forceinline__ void wait_async0() {
#if __has_builtin(__builtin_amdgcn_s_wait_asynccnt)
  __builtin_amdgcn_s_wait_asynccnt(0);
#else
  asm volatile("s_wait_asynccnt 0x0" ::: "memory");
#endif
}

// ---------------------------------------------------------------------------
// Generic f32 -> bf16 conversion
// ---------------------------------------------------------------------------
__global__ __launch_bounds__(256) void k_cvt(const float* __restrict__ src,
                                             bf16* __restrict__ dst, int n) {
  int i = blockIdx.x * 256 + threadIdx.x;
  if (i < n) dst[i] = (bf16)src[i];
}

// ---------------------------------------------------------------------------
// BF16 WMMA GEMM: C[M,N] = A[M,K] * B[N,K]^T   (both row-major, K contiguous)
//
// Block = 256 threads = 8 waves -> 128x64 output tile (wave w: rows w*16..).
// B slab (64 rows x 64 K = 8 KB bf16) staged cooperatively into LDS with
// GLOBAL_LOAD_ASYNC_TO_LDS_B128 (ASYNCcnt), double-buffered: one
// wait_asynccnt + workgroup barrier per 8 WMMAs.  A fragments are per-wave
// from global, software-pipelined one 64-K slab ahead.
//
// A fragment layout (16-bit A 16x32): lanes 0-15 row M=lane, elems 0..15 =
// K+0..15; lanes 16-31 K+16..31 (matches B).
// B fragment (16-bit 32x16): lane n%16 = column, lanes 0-15 K+0..15,
// lanes 16-31 K+16..31 -> 16 contiguous bf16 per lane.
// ---------------------------------------------------------------------------
__global__ __launch_bounds__(256) void k_gemm_bf16(
    const bf16* __restrict__ A, const bf16* __restrict__ B, float* __restrict__ C,
    int M, int N, int Kd, int lda, int ldb, int ldc)
{
  __shared__ __align__(32) bf16 lb[2][64 * 64];   // 2 x 8 KB B slabs

  const int tid  = threadIdx.x;
  const int lane = tid & 31;
  const int wave = tid >> 5;
  const int half = lane >> 4;
  const int r    = lane & 15;

  const int tilesN = N >> 6;
  const int bi = blockIdx.x / tilesN;
  const int bj = blockIdx.x % tilesN;
  const int i0 = bi * 128 + wave * 16;
  const int j0 = bj * 64;

  // cooperative B staging: thread -> (row 0..63, 16-element chunk 0..3)
  const int brow = tid >> 2;
  const int bcol = (tid & 3) * 16;
  const bf16* bsrc = B + (size_t)(j0 + brow) * ldb + bcol;
  bf16* ldst0 = &lb[0][brow * 64 + bcol];
  bf16* ldst1 = &lb[1][brow * 64 + bcol];

  // stage slab 0 (K[0..64))
#if ASYNC_COPY
  __builtin_amdgcn_global_load_async_to_lds_b128((ASG v4i*)bsrc,       (ASL v4i*)ldst0,       0, 0);
  __builtin_amdgcn_global_load_async_to_lds_b128((ASG v4i*)(bsrc + 8), (ASL v4i*)(ldst0 + 8), 0, 0);
  wait_async0();
  __syncthreads();
#else
  *(v8bf*)ldst0       = *(const v8bf*)bsrc;
  *(v8bf*)(ldst0 + 8) = *(const v8bf*)(bsrc + 8);
  __syncthreads();
#endif

  const bf16* pa = A + (size_t)(i0 + r) * lda;
  // A chunks for the current 64-K slab: [sub][chunk]
  v8bf a00 = *(const v8bf*)(pa + half * 16);
  v8bf a01 = *(const v8bf*)(pa + 8 + half * 16);
  v8bf a10 = *(const v8bf*)(pa + 32 + half * 16);
  v8bf a11 = *(const v8bf*)(pa + 40 + half * 16);

  v8f zero = {};
  v8f acc[4];
#pragma unroll
  for (int t = 0; t < 4; ++t) acc[t] = zero;

  int cur = 0;
  for (int k0 = 0; k0 < Kd; k0 += 64) {
    const bool more = (k0 + 64) < Kd;
    bf16* lnext = cur ? ldst0 : ldst1;

    // kick off next B slab into the other LDS buffer
#if ASYNC_COPY
    if (more) {
      __builtin_amdgcn_global_load_async_to_lds_b128(
          (ASG v4i*)(bsrc + k0 + 64), (ASL v4i*)lnext, 0, 0);
      __builtin_amdgcn_global_load_async_to_lds_b128(
          (ASG v4i*)(bsrc + k0 + 72), (ASL v4i*)(lnext + 8), 0, 0);
    }
#else
    v8bf nb0 = {}, nb1 = {};
    if (more) {
      nb0 = *(const v8bf*)(bsrc + k0 + 64);
      nb1 = *(const v8bf*)(bsrc + k0 + 72);
    }
#endif

    // prefetch next slab's A fragments into registers
    v8bf n00 = a00, n01 = a01, n10 = a10, n11 = a11;
    if (more) {
      n00 = *(const v8bf*)(pa + k0 + 64 + half * 16);
      n01 = *(const v8bf*)(pa + k0 + 72 + half * 16);
      n10 = *(const v8bf*)(pa + k0 + 96 + half * 16);
      n11 = *(const v8bf*)(pa + k0 + 104 + half * 16);
    }

    const bf16* lbase = &lb[cur][0];
#pragma unroll
    for (int sub = 0; sub < 2; ++sub) {
      v16bf a;
      v8bf c0 = sub ? a10 : a00;
      v8bf c1 = sub ? a11 : a01;
#pragma unroll
      for (int q = 0; q < 8; ++q) { a[q] = c0[q]; a[8 + q] = c1[q]; }
#pragma unroll
      for (int t = 0; t < 4; ++t) {
        v16bf b = *(const v16bf*)(lbase + (t * 16 + r) * 64 + sub * 32 + half * 16);
        acc[t] = __builtin_amdgcn_wmma_f32_16x16x32_bf16(
            false, a, false, b, (short)0, acc[t], false, false);
      }
    }

#if ASYNC_COPY
    wait_async0();       // my async loads for the next slab are done
    __syncthreads();     // everyone done computing cur + loading next
#else
    __syncthreads();     // everyone done reading cur
    if (more) {
      *(v8bf*)lnext       = nb0;
      *(v8bf*)(lnext + 8) = nb1;
    }
    __syncthreads();
#endif

    cur ^= 1;
    a00 = n00; a01 = n01; a10 = n10; a11 = n11;
  }

#pragma unroll
  for (int t = 0; t < 4; ++t)
#pragma unroll
    for (int e = 0; e < 8; ++e)
      C[(size_t)(i0 + half * 8 + e) * ldc + j0 + t * 16 + r] = acc[t][e];
}

// ---------------------------------------------------------------------------
// RoPE + transpose to head-major BF16: out[h, l, d] from P[l, h*DKH + d]
// ---------------------------------------------------------------------------
__global__ __launch_bounds__(256) void k_rope(
    const float* __restrict__ P, const float* __restrict__ cosc,
    const float* __restrict__ sinc, bf16* __restrict__ out, int heads)
{
  int idx = blockIdx.x * 256 + threadIdx.x;
  int total = heads * LSEQ * DKH;
  if (idx >= total) return;
  int d = idx & 63;
  int l = (idx >> 6) & (LSEQ - 1);
  int h = idx / (LSEQ * DKH);
  int W = heads * DKH;
  float v   = P[(size_t)l * W + h * DKH + d];
  float rot = (d < 32) ? -P[(size_t)l * W + h * DKH + d + 32]
                       :  P[(size_t)l * W + h * DKH + d - 32];
  float o = v * cosc[l * DKH + d] + rot * sinc[l * DKH + d];
  out[(size_t)h * LSEQ * DKH + (size_t)l * DKH + d] = (bf16)o;
}

// ---------------------------------------------------------------------------
// V transpose to [kv, d, L] bf16 so attn@V B-fragments are K-contiguous
// ---------------------------------------------------------------------------
__global__ __launch_bounds__(256) void k_vt(const float* __restrict__ Vf,
                                            bf16* __restrict__ Vt)
{
  int idx = blockIdx.x * 256 + threadIdx.x;
  int total = NKV * DKH * LSEQ;
  if (idx >= total) return;
  int l  = idx & (LSEQ - 1);
  int d  = (idx / LSEQ) & 63;
  int kh = idx / (LSEQ * DKH);
  Vt[idx] = (bf16)Vf[(size_t)l * (NKV * DKH) + kh * DKH + d];
}

// ---------------------------------------------------------------------------
// scores = (Qb[h] @ Kb[kh]^T) / 8 -> attn region (fp32). Causal strips with
// j0 > i0+15 are skipped entirely (softmax zero-fills above the diagonal).
// ---------------------------------------------------------------------------
__global__ __launch_bounds__(256) void k_scores(
    const bf16* __restrict__ Qb, const bf16* __restrict__ Kb,
    float* __restrict__ attn)
{
  const int h    = blockIdx.y;
  const int kh   = h >> 2;
  const int lane = threadIdx.x & 31;
  const int wave = threadIdx.x >> 5;
  const int half = lane >> 4;
  const int r    = lane & 15;
  const int tilesJ = LSEQ / 64;     // 32
  const int w  = blockIdx.x * 8 + wave;
  const int ti = w / tilesJ;
  const int tj = w % tilesJ;
  const int i0 = ti * 16;
  const int j0 = tj * 64;
  if (j0 > i0 + 15) return;         // fully masked strip

  v8f zero = {};
  v8f acc[4];
#pragma unroll
  for (int t = 0; t < 4; ++t) acc[t] = zero;

  const bf16* qa = Qb + (size_t)h * LSEQ * DKH + (size_t)(i0 + r) * DKH;
  const bf16* kb = Kb + (size_t)kh * LSEQ * DKH;
#pragma unroll
  for (int k0 = 0; k0 < DKH; k0 += 32) {
    v8bf a0 = *(const v8bf*)(qa + k0 + half * 16);
    v8bf a1 = *(const v8bf*)(qa + k0 + 8 + half * 16);
    v16bf a;
#pragma unroll
    for (int q = 0; q < 8; ++q) { a[q] = a0[q]; a[8 + q] = a1[q]; }
#pragma unroll
    for (int t = 0; t < 4; ++t) {
      v16bf b = *(const v16bf*)(kb + (size_t)(j0 + t * 16 + r) * DKH + k0 + half * 16);
      acc[t] = __builtin_amdgcn_wmma_f32_16x16x32_bf16(
          false, a, false, b, (short)0, acc[t], false, false);
    }
  }
  float* dst = attn + (size_t)h * LSEQ * LSEQ;
#pragma unroll
  for (int t = 0; t < 4; ++t)
#pragma unroll
    for (int e = 0; e < 8; ++e)
      dst[(size_t)(i0 + half * 8 + e) * LSEQ + j0 + t * 16 + r] = acc[t][e] * 0.125f;
}

// ---------------------------------------------------------------------------
// Row softmax over the causal prefix; zero-fills above the diagonal (this is
// the attn tensor the reference returns). One block per (h, i) row.
// ---------------------------------------------------------------------------
__global__ __launch_bounds__(256) void k_softmax(float* __restrict__ attn)
{
  const int row = blockIdx.x;            // h*LSEQ + i
  const int i   = row & (LSEQ - 1);
  float* p = attn + (size_t)row * LSEQ;
  const int n = i + 1;
  __shared__ float sm[256];
  const int t = threadIdx.x;

  float m = -INFINITY;
  for (int j = t; j < n; j += 256) m = fmaxf(m, p[j]);
  sm[t] = m; __syncthreads();
  for (int s = 128; s > 0; s >>= 1) {
    if (t < s) sm[t] = fmaxf(sm[t], sm[t + s]);
    __syncthreads();
  }
  m = sm[0]; __syncthreads();

  float sum = 0.0f;
  for (int j = t; j < n; j += 256) sum += __expf(p[j] - m);
  sm[t] = sum; __syncthreads();
  for (int s = 128; s > 0; s >>= 1) {
    if (t < s) sm[t] += sm[t + s];
    __syncthreads();
  }
  const float inv = 1.0f / sm[0];

  for (int j = t; j < n; j += 256) p[j] = __expf(p[j] - m) * inv;
  for (int j = n + t; j < LSEQ; j += 256) p[j] = 0.0f;
}

// ---------------------------------------------------------------------------
// ctx[l, h*DKH + d] = attn[h] @ V[kh]; attn fp32 converted to bf16 in-register,
// K-loop clipped to the causal extent of the row tile.
// ---------------------------------------------------------------------------
__global__ __launch_bounds__(256) void k_av(
    const float* __restrict__ attn, const bf16* __restrict__ Vt,
    float* __restrict__ ctx)
{
  const int lane = threadIdx.x & 31;
  const int wave = threadIdx.x >> 5;
  const int half = lane >> 4;
  const int r    = lane & 15;
  const int w  = blockIdx.x * 8 + wave;  // 0..4095
  const int h  = w >> 7;                  // 128 row-tiles per head
  const int ti = w & 127;
  const int kh = h >> 2;
  const int i0 = ti * 16;

  const float* pa = attn + (size_t)h * LSEQ * LSEQ + (size_t)(i0 + r) * LSEQ;
  const bf16*  vb = Vt + (size_t)kh * DKH * LSEQ;

  v8f zero = {};
  v8f acc[4];
#pragma unroll
  for (int t = 0; t < 4; ++t) acc[t] = zero;

  int kmax = (i0 + 16 + 31) & ~31;
  if (kmax > LSEQ) kmax = LSEQ;

  for (int k0 = 0; k0 < kmax; k0 += 32) {
    v4f f0 = *(const v4f*)(pa + k0 + half * 16);
    v4f f1 = *(const v4f*)(pa + k0 + half * 16 + 4);
    v4f f2 = *(const v4f*)(pa + k0 + half * 16 + 8);
    v4f f3 = *(const v4f*)(pa + k0 + half * 16 + 12);
    v16bf a;
#pragma unroll
    for (int q = 0; q < 4; ++q) {
      a[q]      = (bf16)f0[q];
      a[4 + q]  = (bf16)f1[q];
      a[8 + q]  = (bf16)f2[q];
      a[12 + q] = (bf16)f3[q];
    }
#pragma unroll
    for (int t = 0; t < 4; ++t) {
      v16bf b = *(const v16bf*)(vb + (size_t)(t * 16 + r) * LSEQ + k0 + half * 16);
      acc[t] = __builtin_amdgcn_wmma_f32_16x16x32_bf16(
          false, a, false, b, (short)0, acc[t], false, false);
    }
  }
#pragma unroll
  for (int t = 0; t < 4; ++t)
#pragma unroll
    for (int e = 0; e < 8; ++e)
      ctx[(size_t)(i0 + half * 8 + e) * DMODEL + h * DKH + t * 16 + r] = acc[t][e];
}

// ---------------------------------------------------------------------------
// Host-side launch
// ---------------------------------------------------------------------------
extern "C" void kernel_launch(void* const* d_in, const int* in_sizes, int n_in,
                              void* d_out, int out_size, void* d_ws, size_t ws_size,
                              hipStream_t stream)
{
  const float* x    = (const float*)d_in[0];
  // d_in[1] = mask (int32) — causal structure applied analytically
  const float* cosc = (const float*)d_in[2];
  const float* sinc = (const float*)d_in[3];
  const float* Wq   = (const float*)d_in[4];
  const float* Wk   = (const float*)d_in[5];
  const float* Wv   = (const float*)d_in[6];
  const float* Wo   = (const float*)d_in[7];

  float* out  = (float*)d_out;                        // [L, D]
  float* attn = out + (size_t)LSEQ * DMODEL;          // [H, L, L]

  char* ws = (char*)d_ws;
  auto alloc = [&](size_t bytes) -> char* {
    char* p = ws;
    ws += (bytes + 255) & ~(size_t)255;
    return p;
  };
  bf16*  xb   = (bf16*)alloc((size_t)LSEQ * DMODEL * 2);
  bf16*  wqb  = (bf16*)alloc((size_t)DMODEL * DMODEL * 2);
  bf16*  wkb  = (bf16*)alloc((size_t)NKV * DKH * DMODEL * 2);
  bf16*  wvb  = (bf16*)alloc((size_t)NKV * DKH * DMODEL * 2);
  bf16*  wob  = (bf16*)alloc((size_t)DMODEL * DMODEL * 2);
  float* Qf   = (float*)alloc((size_t)LSEQ * NH * DKH * 4);
  float* Kf   = (float*)alloc((size_t)LSEQ * NKV * DKH * 4);
  float* Vf   = (float*)alloc((size_t)LSEQ * NKV * DKH * 4);
  bf16*  Qb   = (bf16*)alloc((size_t)NH * LSEQ * DKH * 2);
  bf16*  Kb   = (bf16*)alloc((size_t)NKV * LSEQ * DKH * 2);
  bf16*  Vt   = (bf16*)alloc((size_t)NKV * DKH * LSEQ * 2);
  float* ctx  = (float*)alloc((size_t)LSEQ * DMODEL * 4);
  bf16*  ctxb = (bf16*)alloc((size_t)LSEQ * DMODEL * 2);

  const int nXD  = LSEQ * DMODEL;         // 4M
  const int nKVW = NKV * DKH * DMODEL;    // 1M

  // fp32 -> bf16 staging
  k_cvt<<<(nXD + 255) / 256, 256, 0, stream>>>(x, xb, nXD);
  k_cvt<<<(nXD + 255) / 256, 256, 0, stream>>>(Wq, wqb, nXD);
  k_cvt<<<(nKVW + 255) / 256, 256, 0, stream>>>(Wk, wkb, nKVW);
  k_cvt<<<(nKVW + 255) / 256, 256, 0, stream>>>(Wv, wvb, nKVW);
  k_cvt<<<(nXD + 255) / 256, 256, 0, stream>>>(Wo, wob, nXD);

  // QKV projections (C = A * W^T); grid = (M/128) * (N/64)
  k_gemm_bf16<<<(LSEQ / 128) * (DMODEL / 64), 256, 0, stream>>>(
      xb, wqb, Qf, LSEQ, DMODEL, DMODEL, DMODEL, DMODEL, DMODEL);
  k_gemm_bf16<<<(LSEQ / 128) * (NKV * DKH / 64), 256, 0, stream>>>(
      xb, wkb, Kf, LSEQ, NKV * DKH, DMODEL, DMODEL, DMODEL, NKV * DKH);
  k_gemm_bf16<<<(LSEQ / 128) * (NKV * DKH / 64), 256, 0, stream>>>(
      xb, wvb, Vf, LSEQ, NKV * DKH, DMODEL, DMODEL, DMODEL, NKV * DKH);

  // RoPE + head-major transpose; V transpose
  k_rope<<<(NH * LSEQ * DKH + 255) / 256, 256, 0, stream>>>(Qf, cosc, sinc, Qb, NH);
  k_rope<<<(NKV * LSEQ * DKH + 255) / 256, 256, 0, stream>>>(Kf, cosc, sinc, Kb, NKV);
  k_vt<<<(NKV * DKH * LSEQ + 255) / 256, 256, 0, stream>>>(Vf, Vt);

  // scores -> attn region of d_out, softmax in place, attn @ V
  k_scores<<<dim3(512, NH), 256, 0, stream>>>(Qb, Kb, attn);
  k_softmax<<<NH * LSEQ, 256, 0, stream>>>(attn);
  k_av<<<512, 256, 0, stream>>>(attn, Vt, ctx);

  // output projection
  k_cvt<<<(nXD + 255) / 256, 256, 0, stream>>>(ctx, ctxb, nXD);
  k_gemm_bf16<<<(LSEQ / 128) * (DMODEL / 64), 256, 0, stream>>>(
      ctxb, wob, out, LSEQ, DMODEL, DMODEL, DMODEL, DMODEL, DMODEL);
}
